// FlashGemma3Attention_9337258901988
// MI455X (gfx1250) — matrix operand order repaired and checked by reading, not compiled, not run on hardware
//
#include <hip/hip_runtime.h>
#include <hip/hip_bf16.h>

// ---------------------------------------------------------------------------
// Gemma3-style windowed attention layer for MI455X (gfx1250, wave32, WMMA).
// Pipeline: f32->f16 convert, QKV GEMM (WMMA + async LDS double-buffering),
// RMSNorm+RoPE, windowed flash attention (WMMA + LDS re-layout),
// output GEMM (WMMA, f32 store).
// ---------------------------------------------------------------------------

typedef __attribute__((ext_vector_type(16))) _Float16 v16h;
typedef __attribute__((ext_vector_type(8)))  _Float16 v8h;
typedef __attribute__((ext_vector_type(4)))  _Float16 v4h;
typedef __attribute__((ext_vector_type(8)))  float    v8f;
typedef __attribute__((ext_vector_type(4)))  float    v4f;
typedef __attribute__((ext_vector_type(4)))  int      v4i;

#define T_LEN 4096
#define HID   2560
#define NH    8
#define NKV   4
#define HD    256
#define QKVW  ((NH + 2 * NKV) * HD)   // 4096 = qkv row width
#define ATTW  (NH * HD)               // 2048
#define ASCALE 0.0625f                // 256^-0.5

// ---------------- async global->LDS copy (CDNA5 ASYNCcnt path) --------------
// Guarded: falls back to plain load + ds_store staging if builtin is absent.
// Builtin signature (from hipcc diagnostic): first param is a pointer to
// int32x4 — i.e. (__global v4i*, __local v4i*, imm offset, imm cpol).

#if __has_builtin(__builtin_amdgcn_global_load_async_to_lds_b128)
#define HAVE_ASYNC_LDS 1
__device__ __forceinline__ void async_cp_b128(const _Float16* g, _Float16* l) {
  // generic global pointer == AS1 address; low 32 bits of a generic LDS
  // pointer are the LDS byte offset (flat aperture encoding).
  auto gp = (__attribute__((address_space(1))) v4i*)(unsigned long long)(uintptr_t)g;
  auto lp = (__attribute__((address_space(3))) v4i*)(unsigned int)(uintptr_t)l;
  __builtin_amdgcn_global_load_async_to_lds_b128(gp, lp, 0, 0);
}
#else
#define HAVE_ASYNC_LDS 0
__device__ __forceinline__ void async_cp_b128(const _Float16* g, _Float16* l) {
  *(v8h*)l = *(const v8h*)g;
}
#endif

__device__ __forceinline__ void async_wait_all() {
#if HAVE_ASYNC_LDS && __has_builtin(__builtin_amdgcn_s_wait_asynccnt)
  __builtin_amdgcn_s_wait_asynccnt(0);
#endif
}

// ---------------- WMMA helpers (lane layouts per CDNA5 ISA 7.12.2) ----------

__device__ __forceinline__ v8f wmma_f16(v16h a, v16h b, v8f c) {
  return __builtin_amdgcn_wmma_f32_16x16x32_f16(
      /*neg_a=*/false, a, /*neg_b=*/false, b,
      /*c_mod=*/(short)0, c, /*reuse_a=*/false, /*reuse_b=*/false);
}

// A-matrix 16x32 f16 frag at (m0,k0) of a row-major MxK matrix (ld halves).
// lane L: M = L%16 ; elems 0-7 -> K = k0 + (L<16?0:8)+e ; elems 8-15 -> +16.
__device__ __forceinline__ v16h load_frag_a(const _Float16* base, int ld,
                                            int m0, int k0, int lane) {
  int m  = m0 + (lane & 15);
  int ko = k0 + ((lane & 16) ? 8 : 0);
  const _Float16* p = base + (size_t)m * ld + ko;
  union { v16h v; v8h h[2]; } u;
  u.h[0] = *(const v8h*)(p);
  u.h[1] = *(const v8h*)(p + 16);
  return u.v;
}

// B-matrix 32x16 f16 frag (logical KxN) stored as NxK row-major (ld halves).
// lane L: N = n0 + L%16 ; elems 0-7 -> K = k0 + (L<16?0:8)+e ; elems 8-15 -> +16.
__device__ __forceinline__ v16h load_frag_b(const _Float16* base, int ld,
                                            int n0, int k0, int lane) {
  int n  = n0 + (lane & 15);
  int ko = k0 + ((lane & 16) ? 8 : 0);
  const _Float16* p = base + (size_t)n * ld + ko;
  union { v16h v; v8h h[2]; } u;
  u.h[0] = *(const v8h*)(p);
  u.h[1] = *(const v8h*)(p + 16);
  return u.v;
}

// ---------------- fp32 -> f16 convert ---------------------------------------

__global__ __launch_bounds__(256) void cvt_f32_f16(const float* __restrict__ src,
                                                   _Float16* __restrict__ dst,
                                                   long n) {
  long i = ((long)blockIdx.x * blockDim.x + threadIdx.x) * 4;
  if (i + 3 < n) {
    v4f v = *(const v4f*)(src + i);
    v4h h;
    h[0] = (_Float16)v[0]; h[1] = (_Float16)v[1];
    h[2] = (_Float16)v[2]; h[3] = (_Float16)v[3];
    *(v4h*)(dst + i) = h;
  } else {
    for (long j = i; j < n; ++j) dst[j] = (_Float16)src[j];
  }
}

// ---------------- WMMA GEMM: C(MxN) = A(MxK) * B(NxK)^T ---------------------
// Block tile 64x256, 8 waves (2 down M x 4 across N), wave tile 32x64.
// A/B K-slabs staged into double-buffered LDS with async copies so the
// next slab's global->LDS DMA overlaps the current slab's 8 WMMAs per wave.

#define BM 64
#define BN 256
#define BK 32

__global__ __launch_bounds__(256) void gemm_wmma(const _Float16* __restrict__ A,
                                                 const _Float16* __restrict__ B,
                                                 _Float16* __restrict__ Ch,
                                                 float* __restrict__ Cf,
                                                 int M, int N, int K) {
  __shared__ __align__(16) _Float16 As[2][BM * BK];   // 4 KB each
  __shared__ __align__(16) _Float16 Bs[2][BN * BK];   // 16 KB each

  const int tid  = threadIdx.x;
  const int lane = tid & 31;
  const int wave = tid >> 5;
  const int wm = wave & 1;          // 2 waves down M (32 rows each)
  const int wn = wave >> 1;         // 4 waves across N (64 cols each)
  const int m0 = blockIdx.y * BM;
  const int n0 = blockIdx.x * BN;

  // staging map: A tile = 64x32 halves -> 1 b128/thread;
  //              B tile = 256x32 halves -> 4 b128/thread (one row each)
  const int ar = tid >> 2, ac = (tid & 3) * 8;
  const int br = tid;

  const _Float16* Ag = A + (size_t)(m0 + ar) * K + ac;
  const _Float16* Bg = B + (size_t)(n0 + br) * K;

  auto stage = [&](int buf, int k0) {
    async_cp_b128(Ag + k0, &As[buf][ar * BK + ac]);
    const _Float16* g = Bg + k0;
    _Float16* l = &Bs[buf][br * BK];
#pragma unroll
    for (int c = 0; c < 4; ++c) async_cp_b128(g + c * 8, l + c * 8);
  };

  v8f acc[2][4] = {};
  const int nK = K / BK;

  stage(0, 0);
  async_wait_all();
  __syncthreads();

  for (int ks = 0; ks < nK; ++ks) {
    const int buf = ks & 1;
    if (ks + 1 < nK) stage(buf ^ 1, (ks + 1) * BK);   // overlap with compute

    v16h a0 = load_frag_a(&As[buf][0], BK, wm * 32,      0, lane);
    v16h a1 = load_frag_a(&As[buf][0], BK, wm * 32 + 16, 0, lane);
#pragma unroll
    for (int t = 0; t < 4; ++t) {
      v16h b = load_frag_b(&Bs[buf][0], BK, wn * 64 + t * 16, 0, lane);
      acc[0][t] = wmma_f16(a0, b, acc[0][t]);
      acc[1][t] = wmma_f16(a1, b, acc[1][t]);
    }

    async_wait_all();
    __syncthreads();
  }

  const int row = (lane & 16) ? 8 : 0;  // C layout: lanes16-31 hold M=r+8
  const int col = lane & 15;
#pragma unroll
  for (int mi = 0; mi < 2; ++mi) {
#pragma unroll
    for (int t = 0; t < 4; ++t) {
#pragma unroll
      for (int r = 0; r < 8; ++r) {
        size_t idx = (size_t)(m0 + wm * 32 + mi * 16 + row + r) * N
                   + (n0 + wn * 64 + t * 16 + col);
        if (Cf) Cf[idx] = acc[mi][t][r];
        else    Ch[idx] = (_Float16)acc[mi][t][r];
      }
    }
  }
}

// ---------------- RMSNorm + RoPE on Q/K heads (in place, f16 qkv) -----------
// grid: (T, NH+NKV). heads [0,NH) use q_norm_w, [NH,NH+NKV) use k_norm_w.

__global__ __launch_bounds__(256) void rmsnorm_rope(_Float16* __restrict__ qkv,
                                                    const float* __restrict__ cosT,
                                                    const float* __restrict__ sinT,
                                                    const float* __restrict__ qw,
                                                    const float* __restrict__ kw) {
  int t = blockIdx.x;
  int h = blockIdx.y;                    // 0..NH+NKV-1
  _Float16* row = qkv + (size_t)t * QKVW + (size_t)h * HD;
  const float* w = (h < NH) ? qw : kw;

  int d = threadIdx.x;                   // 0..255
  float x = (float)row[d];
  float ss = x * x;
#pragma unroll
  for (int m = 16; m >= 1; m >>= 1) ss += __shfl_xor(ss, m, 32);

  __shared__ float red[8];
  __shared__ float ybuf[HD];
  if ((threadIdx.x & 31) == 0) red[threadIdx.x >> 5] = ss;
  __syncthreads();
  float var = 0.f;
#pragma unroll
  for (int i = 0; i < 8; ++i) var += red[i];
  var *= (1.0f / (float)HD);
  float y = x * rsqrtf(var + 1e-6f) * w[d];

  ybuf[d] = y;
  __syncthreads();

  float out;
  if (d < HD / 2) {
    float c = cosT[t * (HD / 2) + d], s = sinT[t * (HD / 2) + d];
    out = ybuf[d] * c - ybuf[d + HD / 2] * s;
  } else {
    int dd = d - HD / 2;
    float c = cosT[t * (HD / 2) + dd], s = sinT[t * (HD / 2) + dd];
    out = ybuf[d] * c + ybuf[dd] * s;
  }
  row[d] = (_Float16)out;
}

// ---------------- Windowed flash attention, 1 wave per (head, 16-row tile) --
// S = Q K^T via WMMA (K-dim 256 = 8 steps), online softmax in f32,
// P re-laid out through LDS into A-frags, V staged transposed in LDS,
// O (16x256) accumulated as 16 WMMA C-frags.

__global__ __launch_bounds__(32) void attn_wmma(const _Float16* __restrict__ qkv,
                                                _Float16* __restrict__ outh,
                                                const int* __restrict__ window_p) {
  const int window = *window_p;
  const int h   = blockIdx.y;                 // 0..NH-1
  const int i0  = blockIdx.x * 16;            // q rows [i0, i0+16)
  const int kvh = h / (NH / NKV);
  const int lane = threadIdx.x & 31;
  const int col  = lane & 15;
  const int rbase = (lane & 16) ? 8 : 0;

  const _Float16* Qb = qkv + (size_t)h * HD;
  const _Float16* Kb = qkv + (size_t)(NH + kvh) * HD;
  const _Float16* Vb = qkv + (size_t)(NH + NKV + kvh) * HD;

  __shared__ __align__(16) _Float16 Vt[HD * 32];   // V^T tile: Vt[d*32 + kc]
  __shared__ __align__(16) _Float16 Pl[16 * 32];   // P tile (row-major 16x32)

  // Q fragments for the whole head dim (8 x 16x32)
  v16h qf[8];
#pragma unroll
  for (int kk = 0; kk < 8; ++kk)
    qf[kk] = load_frag_a(Qb, QKVW, i0, kk * 32, lane);

  v8f o[16] = {};
  float mr[8], lr[8];
#pragma unroll
  for (int r = 0; r < 8; ++r) { mr[r] = -3.0e38f; lr[r] = 0.f; }

  int jlo = i0 - window; if (jlo < 0) jlo = 0;
  int jstart = (jlo / 32) * 32;
  int jend = i0 + 16;                          // keys j satisfy j <= i < i0+16

  for (int jb = jstart; jb < jend; jb += 32) {
    // near-cache prefetch of the next key block's K/V rows
    {
      int jn = jb + 32 + lane;
      if (jn < T_LEN) {
        __builtin_prefetch(Kb + (size_t)jn * QKVW, 0, 3);
        __builtin_prefetch(Vb + (size_t)jn * QKVW, 0, 3);
      }
    }

    // ---- stage V[jb..jb+31][0..255] transposed into LDS ----
    {
      int dlo = lane * 8;                      // 32 lanes * 8 = 256 d values
#pragma unroll 4
      for (int kc = 0; kc < 32; ++kc) {
        v8h vv = *(const v8h*)(Vb + (size_t)(jb + kc) * QKVW + dlo);
#pragma unroll
        for (int e = 0; e < 8; ++e) Vt[(dlo + e) * 32 + kc] = vv[e];
      }
    }
    __syncthreads();

    // ---- scores: two 16x16 tiles (keys jb..+15, jb+16..+31) ----
    v8f s0 = {}, s1 = {};
#pragma unroll
    for (int kk = 0; kk < 8; ++kk) {
      v16h b0 = load_frag_b(Kb, QKVW, jb,      kk * 32, lane);
      v16h b1 = load_frag_b(Kb, QKVW, jb + 16, kk * 32, lane);
      s0 = wmma_f16(qf[kk], b0, s0);
      s1 = wmma_f16(qf[kk], b1, s1);
    }

    // ---- mask + online softmax (rows live across 16-lane groups) ----
    float corrv[8];
#pragma unroll
    for (int r = 0; r < 8; ++r) {
      int i  = i0 + rbase + r;
      int j0 = jb + col, j1 = jb + 16 + col;
      float a0 = s0[r] * ASCALE;
      float a1 = s1[r] * ASCALE;
      if (!((j0 <= i) && (i - j0 <= window))) a0 = -1.0e30f;
      if (!((j1 <= i) && (i - j1 <= window))) a1 = -1.0e30f;

      float mt = fmaxf(a0, a1);
#pragma unroll
      for (int off = 1; off < 16; off <<= 1) mt = fmaxf(mt, __shfl_xor(mt, off, 32));

      float mnew = fmaxf(mr[r], mt);
      float corr = __expf(mr[r] - mnew);
      float p0 = __expf(a0 - mnew);
      float p1 = __expf(a1 - mnew);
      float rs = p0 + p1;
#pragma unroll
      for (int off = 1; off < 16; off <<= 1) rs += __shfl_xor(rs, off, 32);

      lr[r] = lr[r] * corr + rs;
      mr[r] = mnew;
      corrv[r] = corr;

      Pl[(rbase + r) * 32 + col]      = (_Float16)p0;
      Pl[(rbase + r) * 32 + 16 + col] = (_Float16)p1;
    }

    // rescale accumulated O by per-row correction
#pragma unroll
    for (int tt = 0; tt < 16; ++tt)
#pragma unroll
      for (int r = 0; r < 8; ++r) o[tt][r] = o[tt][r] * corrv[r];

    __syncthreads();

    // ---- P (A-frag from LDS) x V (B-frags from transposed LDS) ----
    v16h pf;
    {
      int m  = lane & 15;
      int ko = (lane & 16) ? 8 : 0;
      union { v16h v; v8h h[2]; } u;
      u.h[0] = *(const v8h*)&Pl[m * 32 + ko];
      u.h[1] = *(const v8h*)&Pl[m * 32 + 16 + ko];
      pf = u.v;
    }
#pragma unroll
    for (int tt = 0; tt < 16; ++tt) {
      int n  = tt * 16 + col;                 // output d index
      int vo = (lane & 16) ? 8 : 0;
      union { v16h v; v8h h[2]; } u;
      u.h[0] = *(const v8h*)&Vt[n * 32 + vo];
      u.h[1] = *(const v8h*)&Vt[n * 32 + vo + 16];
      o[tt] = wmma_f16(pf, u.v, o[tt]);
    }
    __syncthreads();                           // before Vt is overwritten
  }

  // ---- epilogue: divide by row sums, store f16 attn output ----
#pragma unroll
  for (int tt = 0; tt < 16; ++tt) {
#pragma unroll
    for (int r = 0; r < 8; ++r) {
      int t = i0 + rbase + r;
      int d = tt * 16 + col;
      outh[(size_t)t * ATTW + (size_t)h * HD + d] = (_Float16)(o[tt][r] / lr[r]);
    }
  }
}

// ---------------- host-side launch ------------------------------------------

extern "C" void kernel_launch(void* const* d_in, const int* in_sizes, int n_in,
                              void* d_out, int out_size, void* d_ws, size_t ws_size,
                              hipStream_t stream) {
  (void)in_sizes; (void)n_in; (void)out_size; (void)ws_size;
  const float* hidden = (const float*)d_in[0];
  const float* cosT   = (const float*)d_in[1];
  const float* sinT   = (const float*)d_in[2];
  const float* w_qkv  = (const float*)d_in[3];
  const float* w_o    = (const float*)d_in[4];
  const float* q_norm = (const float*)d_in[5];
  const float* k_norm = (const float*)d_in[6];
  const int*   window = (const int*)d_in[7];

  char* ws = (char*)d_ws;
  _Float16* hid_h  = (_Float16*)(ws);                                   // T x HID
  _Float16* wqkv_h = (_Float16*)(ws + (size_t)T_LEN * HID * 2);         // QKVW x HID
  _Float16* wo_h   = (_Float16*)(ws + (size_t)(T_LEN + QKVW) * HID * 2);          // HID x ATTW
  _Float16* qkv_h  = (_Float16*)(ws + (size_t)(T_LEN + QKVW) * HID * 2
                                     + (size_t)HID * ATTW * 2);         // T x QKVW
  _Float16* attn_h = (_Float16*)(ws + (size_t)(T_LEN + QKVW) * HID * 2
                                     + (size_t)HID * ATTW * 2
                                     + (size_t)T_LEN * QKVW * 2);       // T x ATTW

  // fp32 -> f16 staging
  {
    long n0 = (long)T_LEN * HID;
    long n1 = (long)QKVW * HID;
    long n2 = (long)HID * ATTW;
    cvt_f32_f16<<<dim3((unsigned)((n0 / 4 + 255) / 256)), 256, 0, stream>>>(hidden, hid_h, n0);
    cvt_f32_f16<<<dim3((unsigned)((n1 / 4 + 255) / 256)), 256, 0, stream>>>(w_qkv, wqkv_h, n1);
    cvt_f32_f16<<<dim3((unsigned)((n2 / 4 + 255) / 256)), 256, 0, stream>>>(w_o, wo_h, n2);
  }

  // QKV projection: (T x HID) * (QKVW x HID)^T -> T x QKVW (f16)
  gemm_wmma<<<dim3(QKVW / BN, T_LEN / BM), 256, 0, stream>>>(
      hid_h, wqkv_h, qkv_h, nullptr, T_LEN, QKVW, HID);

  // RMSNorm + RoPE on Q and K heads (in place)
  rmsnorm_rope<<<dim3(T_LEN, NH + NKV), 256, 0, stream>>>(
      qkv_h, cosT, sinT, q_norm, k_norm);

  // Windowed flash attention -> T x ATTW (f16)
  attn_wmma<<<dim3(T_LEN / 16, NH), 32, 0, stream>>>(qkv_h, attn_h, window);

  // Output projection: (T x ATTW) * (HID x ATTW)^T -> T x HID (f32 to d_out)
  gemm_wmma<<<dim3(HID / BN, T_LEN / BM), 256, 0, stream>>>(
      attn_h, wo_h, nullptr, (float*)d_out, T_LEN, HID, ATTW);
}
